// GRUDecoder_37288906063905
// MI455X (gfx1250) — matrix operand order repaired and checked by reading, not compiled, not run on hardware
//
#include <hip/hip_runtime.h>
#include <hip/hip_bf16.h>

// ---------------------------------------------------------------------------
// GRU decoder for MI455X (gfx1250, wave32, WMMA).
//   N=4096 rows, H=512, C=512, E=64, V=256, T=32.
// Strategy:
//   * one-time: ctx->bf16, pack W_hh / W_ih_ctx / fc_W slices into per-lane
//     WMMA B-fragment layout; emb tables (emb @ W) [256 x 1536] / [256 x 256];
//     gx_ctx = ctx @ W_ih[:,64:].T + b_ih ; logit_ctx = ctx @ fc_W[:,64:576].T + fc_b
//   * recurrent kernel: 256 blocks x 16 rows, 8 waves. Per step:
//       gh  = h @ W_hh.T     (12 wmma-tiles/wave, K=512, batched B loads)
//       gates in accumulator layout -> h_new (regs + bf16 LDS tile)
//       logits = logit_ctx + emb_fc[tok] + h_new @ fc_Wh.T (2 tiles/wave)
//       argmax -> next token.  No grid sync needed (rows independent).
//   * anti-LICM: all per-step weight/state loads use ONE per-lane base pointer
//     plus compile-time immediate offsets (24-bit IOFFSET), and an opaque
//     zero-valued SGPR index redefined per iteration keeps them inside the
//     T-loop.  (r2: hoisted fragments spilled to scratch; r3: "+s" on the
//     pointer killed global-AS inference -> flat_load + hoisted address
//     spills.  Opaque *index* keeps the pointer chain clean -> global_load.)
//   * logits stores are non-temporal: 128 MB write-once stream must not evict
//     the L2-resident per-step state (packed weights + gx_ctx, ~32 MB).
// ---------------------------------------------------------------------------

typedef __bf16 bf16_t;
typedef __attribute__((ext_vector_type(16))) __bf16 v16bf;
typedef __attribute__((ext_vector_type(8)))  __bf16 v8bf;
typedef __attribute__((ext_vector_type(8)))  float   v8f;

#define H_DIM 512
#define C_DIM 512
#define V_DIM 256
#define G3    1536
#define KT_N  16          // 512 / 32 k-tiles
#define LDH   520         // LDS h-tile row stride (halves): 512+8 kills bank conflicts
#define LDL   272         // LDS logits row stride (floats)

static __device__ __forceinline__ v8f zero8() {
  v8f z;
#pragma unroll
  for (int i = 0; i < 8; ++i) z[i] = 0.f;
  return z;
}

static __device__ __forceinline__ v16bf make_frag(const bf16_t* lo, const bf16_t* hi) {
  v8bf l = *reinterpret_cast<const v8bf*>(lo);
  v8bf h = *reinterpret_cast<const v8bf*>(hi);
  v16bf r;
#pragma unroll
  for (int i = 0; i < 8; ++i) { r[i] = l[i]; r[8 + i] = h[i]; }
  return r;
}

// Fragment at p: 16 halves (two b128 loads at p and p+8).
static __device__ __forceinline__ v16bf frag_at(const bf16_t* p) {
  return make_frag(p, p + 8);
}

static __device__ __forceinline__ v8f wmma_bf16(v16bf a, v16bf b, v8f c) {
  return __builtin_amdgcn_wmma_f32_16x16x32_bf16(false, a, false, b, (short)0, c,
                                                 false, false);
}

// B fragment: packed [ntile][ktile][lane][16 halves]; lane = column n (mod 16)
// + k-half select, 16 consecutive K values per lane -> two b128 loads.
static __device__ __forceinline__ v16bf load_bpk(const bf16_t* base,
                                                 int nt, int kt, int lane) {
  const bf16_t* p = base + (((size_t)nt * KT_N + kt) * 32 + lane) * 16;
  return make_frag(p, p + 8);
}

// ---------------------------------------------------------------------------
// One-time prep kernels
// ---------------------------------------------------------------------------

__global__ void cvt_ctx_kernel(const float* __restrict__ src,
                               bf16_t* __restrict__ dst, int total) {
  int gid = blockIdx.x * blockDim.x + threadIdx.x;
  if (gid < total) dst[gid] = (bf16_t)src[gid];
}

// Pack rows [n, col_off + k] of a row-major f32 matrix into the WMMA
// B-fragment bf16 layout (K=512 fixed). gid = (nt*16 + kt)*32 + lane.
__global__ void pack_b_kernel(const float* __restrict__ src, int src_ld,
                              int col_off, bf16_t* __restrict__ dst, int ntiles) {
  int gid = blockIdx.x * blockDim.x + threadIdx.x;
  int total = ntiles * KT_N * 32;
  if (gid >= total) return;
  int lane = gid & 31;
  int kt   = (gid >> 5) & 15;
  int n    = (gid >> 9) * 16 + (lane & 15);
  int kb   = kt * 32 + 16 * (lane >> 4);
  const float* s = src + (size_t)n * src_ld + col_off + kb;
  bf16_t* d = dst + (size_t)gid * 16;
#pragma unroll
  for (int i = 0; i < 16; ++i) d[i] = (bf16_t)s[i];
}

// out[v][j] = sum_{e<64} emb_W[v][e] * W[j][e]   (emb contribution tables)
__global__ void emb_tab_kernel(const float* __restrict__ embW,
                               const float* __restrict__ W, int ld, int ncols,
                               float* __restrict__ out) {
  int gid = blockIdx.x * blockDim.x + threadIdx.x;
  if (gid >= 256 * ncols) return;
  int v = gid / ncols, j = gid - v * ncols;
  const float* a = embW + v * 64;
  const float* b = W + (size_t)j * ld;
  float s = 0.f;
#pragma unroll
  for (int e = 0; e < 64; ++e) s += a[e] * b[e];
  out[gid] = s;
}

// out[row][col] = ctx_bf16[row] . Bpacked[:,col] + bias[col]
// 16 rows/block, 8 waves, NTPW n-tiles per wave (templated so the NTPW
// accumulators stay in registers -> NTPW independent WMMA chains for ILP).
template <int NTPW>
__global__ __launch_bounds__(256) void ctx_gemm_kernel(
    const bf16_t* __restrict__ ctxb, const bf16_t* __restrict__ Bp,
    const float* __restrict__ bias, float* __restrict__ outp, int ncols) {
  const int tid = threadIdx.x, lane = tid & 31, wave = tid >> 5;
  const int mlo = lane & 15, kh = lane >> 4;
  const int rb = blockIdx.x << 4;

  v16bf a[16];
  const bf16_t* arow = ctxb + (size_t)(rb + mlo) * C_DIM;
#pragma unroll
  for (int kt = 0; kt < 16; ++kt)
    a[kt] = make_frag(arow + kt * 32 + 16 * kh, arow + kt * 32 + 16 * kh + 8);

  v8f acc[NTPW];
#pragma unroll
  for (int j = 0; j < NTPW; ++j) acc[j] = zero8();

#pragma unroll
  for (int kt = 0; kt < 16; ++kt) {
    v16bf b[NTPW];
#pragma unroll
    for (int j = 0; j < NTPW; ++j)
      b[j] = load_bpk(Bp, wave * NTPW + j, kt, lane);
#pragma unroll
    for (int j = 0; j < NTPW; ++j)
      acc[j] = wmma_bf16(a[kt], b[j], acc[j]);
  }

#pragma unroll
  for (int j = 0; j < NTPW; ++j) {
    int col = (wave * NTPW + j) * 16 + mlo;
    float bv = bias[col];
#pragma unroll
    for (int v = 0; v < 8; ++v)
      outp[(size_t)(rb + v + 8 * kh) * ncols + col] = acc[j][v] + bv;
  }
}

// ---------------------------------------------------------------------------
// Recurrent kernel: each block owns 16 rows for all T steps.
// ---------------------------------------------------------------------------

__global__ __launch_bounds__(256) void gru_step_kernel(
    const int* __restrict__ init_tok, const int* __restrict__ Tp,
    const float* __restrict__ b_hh, const bf16_t* __restrict__ Bhh,
    const bf16_t* __restrict__ Bfch, const float* __restrict__ gx_ctx,
    const float* __restrict__ logit_ctx, const float* __restrict__ emb_gx,
    const float* __restrict__ emb_fc, float* __restrict__ out) {
  __shared__ bf16_t hT[16 * LDH];     // h_new tile (A operand for WMMA)
  __shared__ float  lg[16 * LDL];     // logits tile for argmax
  __shared__ float  cval[256];
  __shared__ int    cidx[256];
  __shared__ int    tokS[16];

  const int tid = threadIdx.x;
  const int lane = tid & 31;
  const int wave = tid >> 5;
  const int mlo = lane & 15;          // C/D: column within tile; A: row
  const int kh = lane >> 4;           // half-select
  const int rb = blockIdx.x << 4;
  const int T = Tp[0];

  if (tid < 16) tokS[tid] = init_tok[rb + tid];

  // h0 = 0: zero A fragments + f32 h state (acc layout, wave owns 64 hidden ch)
  v16bf a[16];
#pragma unroll
  for (int kt = 0; kt < 16; ++kt)
#pragma unroll
    for (int i = 0; i < 16; ++i) a[kt][i] = (bf16_t)0.f;
  v8f hst[4];
#pragma unroll
  for (int i = 0; i < 4; ++i) hst[i] = zero8();

  float bh[3][4];
#pragma unroll
  for (int g = 0; g < 3; ++g)
#pragma unroll
    for (int i = 0; i < 4; ++i)
      bh[g][i] = b_hh[g * H_DIM + wave * 64 + i * 16 + mlo];

  __syncthreads();

  for (int t = 0; t < T; ++t) {
    // Opaque zero index, redefined per iteration ("+s": one SGPR pair).
    // Added into array indices below: load addresses formally depend on it so
    // LICM cannot hoist the t-invariant loads (=> no scratch spills), while
    // the pointer bases stay traceable to kernargs (=> global_load, saddr/imm).
    size_t z = 0;
    asm volatile("" : "+s"(z));

    // Single per-lane fragment base + compile-time immediate offsets
    // (g*524288 + i*16384 + kt*1024 bytes, all < 2^23 -> 24-bit IOFFSET).
    const bf16_t* bbh = Bhh + (size_t)wave * 32768 + (size_t)lane * 16 + z;
    const bf16_t* bfc = Bfch + (size_t)wave * 16384 + (size_t)lane * 16 + z;

    // ---- gh = h @ W_hh.T : wave owns cols {g*512 + wave*64 + i*16 + c} ----
    // Batch the 24 b128 B-loads of each k-tile ahead of its 12 WMMAs so the
    // next k-tile's loads issue underneath the current WMMA block.
    v8f acc[3][4];
#pragma unroll
    for (int g = 0; g < 3; ++g)
#pragma unroll
      for (int i = 0; i < 4; ++i) acc[g][i] = zero8();
#pragma unroll
    for (int kt = 0; kt < 16; ++kt) {
      v16bf b[12];
#pragma unroll
      for (int g = 0; g < 3; ++g)
#pragma unroll
        for (int i = 0; i < 4; ++i)
          b[g * 4 + i] = frag_at(bbh + g * 262144 + i * 8192 + kt * 512);
#pragma unroll
      for (int g = 0; g < 3; ++g)
#pragma unroll
        for (int i = 0; i < 4; ++i)
          acc[g][i] = wmma_bf16(a[kt], b[g * 4 + i], acc[g][i]);
    }

    int trow[8];
#pragma unroll
    for (int v = 0; v < 8; ++v) trow[v] = tokS[v + 8 * kh];

    // ---- gates in accumulator layout: (row = v+8*kh, hidden col) ----
    v8f hnew[4];
#pragma unroll
    for (int i = 0; i < 4; ++i) {
      int colh = wave * 64 + i * 16 + mlo;
#pragma unroll
      for (int v = 0; v < 8; ++v) {
        int row = rb + v + 8 * kh;
        size_t gb = (size_t)row * G3 + z;        // z: keep loads in-loop
        size_t eb = (size_t)trow[v] * G3;
        float gxr = gx_ctx[gb + colh]             + emb_gx[eb + colh];
        float gxz = gx_ctx[gb + H_DIM + colh]     + emb_gx[eb + H_DIM + colh];
        float gxn = gx_ctx[gb + 2 * H_DIM + colh] + emb_gx[eb + 2 * H_DIM + colh];
        float ghr = acc[0][i][v] + bh[0][i];
        float ghz = acc[1][i][v] + bh[1][i];
        float ghn = acc[2][i][v] + bh[2][i];
        float r = 1.f / (1.f + __expf(-(gxr + ghr)));
        float zz = 1.f / (1.f + __expf(-(gxz + ghz)));
        float nn = tanhf(gxn + r * ghn);
        hnew[i][v] = (1.f - zz) * nn + zz * hst[i][v];
      }
    }

    __syncthreads();  // previous readers of hT / tokS are done
#pragma unroll
    for (int i = 0; i < 4; ++i) {
      int colh = wave * 64 + i * 16 + mlo;
#pragma unroll
      for (int v = 0; v < 8; ++v)
        hT[(v + 8 * kh) * LDH + colh] = (bf16_t)hnew[i][v];
      hst[i] = hnew[i];
    }
    __syncthreads();  // hT (h_new) ready

    // Reload A fragments (feeds fc GEMM now and next step's gh)
    {
      const bf16_t* arow = hT + mlo * LDH;
#pragma unroll
      for (int kt = 0; kt < 16; ++kt)
        a[kt] = make_frag(arow + kt * 32 + 16 * kh,
                          arow + kt * 32 + 16 * kh + 8);
    }

    // ---- logits = logit_ctx + emb_fc[tok] + h_new @ fc_Wh.T ----
    v8f accf[2];
#pragma unroll
    for (int j = 0; j < 2; ++j) accf[j] = zero8();
#pragma unroll
    for (int kt = 0; kt < 16; ++kt) {
      v16bf b[2];
#pragma unroll
      for (int j = 0; j < 2; ++j)
        b[j] = frag_at(bfc + j * 8192 + kt * 512);
#pragma unroll
      for (int j = 0; j < 2; ++j)
        accf[j] = wmma_bf16(a[kt], b[j], accf[j]);
    }

#pragma unroll
    for (int j = 0; j < 2; ++j) {
      int colv = (wave * 2 + j) * 16 + mlo;
#pragma unroll
      for (int v = 0; v < 8; ++v) {
        int row = rb + v + 8 * kh;
        float lv = accf[j][v] + logit_ctx[(size_t)row * V_DIM + colv + z] +
                   emb_fc[(size_t)trow[v] * V_DIM + colv];
        // Write-once 128MB stream: keep it out of L2 (preserve resident state).
        __builtin_nontemporal_store(lv, &out[((size_t)row * T + t) * V_DIM + colv]);
        lg[(v + 8 * kh) * LDL + colv] = lv;
      }
    }
    __syncthreads();  // lg ready

    // ---- argmax, first-max semantics (matches jnp.argmax) ----
    {
      int row = tid >> 4, ch = tid & 15, base = ch * 16;
      float best = -3.4e38f; int bi = 0;
#pragma unroll
      for (int q = 0; q < 16; ++q) {
        float x = lg[row * LDL + base + q];
        if (x > best) { best = x; bi = base + q; }
      }
      cval[tid] = best; cidx[tid] = bi;
    }
    __syncthreads();
    if (tid < 16) {
      float best = -3.4e38f; int bi = 0;
#pragma unroll
      for (int c = 0; c < 16; ++c) {           // ascending chunks -> lowest idx tie
        float x = cval[tid * 16 + c];
        if (x > best) { best = x; bi = cidx[tid * 16 + c]; }
      }
      tokS[tid] = bi;
    }
    __syncthreads();  // tokS ready for next step
  }
}

// ---------------------------------------------------------------------------
// Launch
// ---------------------------------------------------------------------------

extern "C" void kernel_launch(void* const* d_in, const int* in_sizes, int n_in,
                              void* d_out, int out_size, void* d_ws,
                              size_t ws_size, hipStream_t stream) {
  const float* encoded  = (const float*)d_in[0];
  const int*   init_tok = (const int*)d_in[1];
  const float* emb_W    = (const float*)d_in[2];
  const float* W_ih     = (const float*)d_in[3];
  const float* W_hh     = (const float*)d_in[4];
  const float* b_ih     = (const float*)d_in[5];
  const float* b_hh     = (const float*)d_in[6];
  const float* fc_W     = (const float*)d_in[7];
  const float* fc_b     = (const float*)d_in[8];
  const int*   Tp       = (const int*)d_in[9];
  float* out = (float*)d_out;

  // Workspace layout (bytes), ~37.3 MB total, all offsets 256B aligned.
  char* w = (char*)d_ws;
  bf16_t* ctxb  = (bf16_t*)(w + 0);          // 4096*512  bf16 :  4.0 MB
  bf16_t* Bhh   = (bf16_t*)(w + 4194304);    // 1536*512  bf16 :  1.5 MB
  bf16_t* Bihc  = (bf16_t*)(w + 5767168);    // 1536*512  bf16 :  1.5 MB
  bf16_t* Bfch  = (bf16_t*)(w + 7340032);    //  256*512  bf16 : 0.25 MB
  bf16_t* Bfcc  = (bf16_t*)(w + 7602176);    //  256*512  bf16 : 0.25 MB
  float*  embgx = (float*)(w + 7864320);     //  256*1536 f32  :  1.5 MB
  float*  embfc = (float*)(w + 9437184);     //  256*256  f32  : 0.25 MB
  float*  gxctx = (float*)(w + 9699328);     // 4096*1536 f32  : 24.0 MB
  float*  lctx  = (float*)(w + 34865152);    // 4096*256  f32  :  4.0 MB

  cvt_ctx_kernel<<<8192, 256, 0, stream>>>(encoded, ctxb, 4096 * 512);
  pack_b_kernel<<<192, 256, 0, stream>>>(W_hh, 512, 0, Bhh, 96);
  pack_b_kernel<<<192, 256, 0, stream>>>(W_ih, 576, 64, Bihc, 96);
  pack_b_kernel<<<32, 256, 0, stream>>>(fc_W, 1088, 576, Bfch, 16);
  pack_b_kernel<<<32, 256, 0, stream>>>(fc_W, 1088, 64, Bfcc, 16);
  emb_tab_kernel<<<1536, 256, 0, stream>>>(emb_W, W_ih, 576, 1536, embgx);
  emb_tab_kernel<<<256, 256, 0, stream>>>(emb_W, fc_W, 1088, 256, embfc);
  ctx_gemm_kernel<12><<<256, 256, 0, stream>>>(ctxb, Bihc, b_ih, gxctx, 1536);
  ctx_gemm_kernel<2><<<256, 256, 0, stream>>>(ctxb, Bfcc, fc_b, lctx, 256);
  gru_step_kernel<<<256, 256, 0, stream>>>(init_tok, Tp, b_hh, Bhh, Bfch, gxctx,
                                           lctx, embgx, embfc, out);
}
